// BranchingLayer_36833639531316
// MI455X (gfx1250) — matrix operand order repaired and checked by reading, not compiled, not run on hardware
//
#include <hip/hip_runtime.h>

typedef _Float16 f16;
typedef __attribute__((ext_vector_type(16))) _Float16 v16h;
typedef __attribute__((ext_vector_type(8)))  _Float16 h8;
typedef __attribute__((ext_vector_type(8)))  float    v8f;
typedef __attribute__((ext_vector_type(4)))  float    f4;

#define E_   32768
#define F_   64
#define G_   64
#define B_   2
#define H_   256
#define P_   8
#define D_   4
#define N0_  (15 * E_)          // 491520 existing node rows
#define TM_  128                // M rows per block
#define ROWS_ (P_ * E_)         // 262144 MLP rows

static __device__ __forceinline__ v16h ld_frag_a(const f16* p) {
  // A-operand pattern: halves {k..k+7} and {k+16..k+23} (ISA 16-bit A 16x32)
  h8 lo = *(const h8*)p, hi = *(const h8*)(p + 16);
  v16h r;
#pragma unroll
  for (int j = 0; j < 8; ++j) { r[j] = lo[j]; r[8 + j] = hi[j]; }
  return r;
}
static __device__ __forceinline__ v16h ld_frag_b(const f16* p) {
  // Contiguous-k pattern: 16 contiguous k per half-wave (unused; B operands use ld_frag_a)
  h8 lo = *(const h8*)p, hi = *(const h8*)(p + 8);
  v16h r;
#pragma unroll
  for (int j = 0; j < 8; ++j) { r[j] = lo[j]; r[8 + j] = hi[j]; }
  return r;
}
#define WMMA_(A, Bv, C) __builtin_amdgcn_wmma_f32_16x16x32_f16( \
    false, (A), false, (Bv), (short)0, (C), false, false)

// ---------------------------------------------------------------------------
// Fused 2-layer MLP: out[m,0:128] = relu([x|g] @ W1 + b1) @ W2 + b2
// Block: 256 threads (8 wave32), M-tile = 128 rows.
// LDS (160KB): [0,64K) W1^T (f16,[n256][k128]) -> reused as H ([m128][k256])
//              [64K,128K) W2^T (f16,[n128][k256])
//              [128K,160K) A tile (f16,[m128][k128])
// Stage 1 computes H^T tiles (A-op = W1^T, B-op = A) so each lane's 8
// accumulator elements are 8 consecutive hidden indices of one m-row ->
// packed f16 writeback with a single ds_store_b128 per tile, directly in the
// row-major [m][k] layout stage 2 reads back with contiguous b128 loads.
// ---------------------------------------------------------------------------
__launch_bounds__(256)
__global__ void mlp_fused_kernel(const float* __restrict__ x,
                                 const float* __restrict__ gf,
                                 const float* __restrict__ W1g,
                                 const float* __restrict__ b1g,
                                 const float* __restrict__ W2g,
                                 const float* __restrict__ b2g,
                                 float* __restrict__ out /* x_new + N0*F */) {
  __shared__ __align__(16) f16 smem[81920];          // 160 KB
  f16* W1t = smem;                                   // 32768 f16
  f16* W2t = smem + 32768;                           // 32768 f16
  f16* At  = smem + 65536;                           // 16384 f16
  f16* Ht  = smem;                                   // reuses W1t after stage 1

  const int  tid  = threadIdx.x;
  const int  lane = tid & 31;
  const int  wave = tid >> 5;
  const int  rc   = lane & 15;           // row (A-op) / col (B-op, C/D)
  const int  hw   = lane >> 4;           // half-wave 0/1
  const long blockRow = (long)blockIdx.x * TM_;

  // --- stage weights (transposed, f16) into LDS ---
  for (int i = tid; i < 256 * 128; i += 256) {       // W1t[n*128+k] = W1[k*256+n]
    int n = i >> 7, k = i & 127;
    W1t[i] = (f16)W1g[k * 256 + n];
  }
  for (int i = tid; i < 128 * 256; i += 256) {       // W2t[n*256+k] = W2[k*128+n]
    int n = i >> 8, k = i & 255;
    W2t[i] = (f16)W2g[k * 128 + n];
  }
  // --- stage A tile: row r = [ x[7E + blockRow + r] | gf[(blockRow+r)%E] ] ---
  {
    int r = tid >> 1;
    int part = tid & 1;
    long m = blockRow + r;
    const float* src = part ? (gf + (m & (E_ - 1)) * 64)
                            : (x + (7L * E_ + m) * 64);
    f16* dst = At + r * 128 + part * 64;
#pragma unroll
    for (int j = 0; j < 64; j += 4) {
      f4 v = *(const f4*)(src + j);
      dst[j + 0] = (f16)v.x; dst[j + 1] = (f16)v.y;
      dst[j + 2] = (f16)v.z; dst[j + 3] = (f16)v.w;
    }
  }
  __syncthreads();

  // ===== Stage 1: H^T tiles.  D[n][m] = sum_k W1t[n][k] * At[m][k]
  // wave owns n-tiles {2w, 2w+1}; m-tiles 0..7.  acc1[mt*2+nt].
  v8f acc1[16];
#pragma unroll
  for (int i = 0; i < 16; ++i) acc1[i] = (v8f){};

  const f16* W1row0 = W1t + ((wave * 2 + 0) * 16 + rc) * 128;
  const f16* W1row1 = W1t + ((wave * 2 + 1) * 16 + rc) * 128;

  v16h a0 = ld_frag_a(W1row0 + hw * 8);
  v16h a1 = ld_frag_a(W1row1 + hw * 8);
  v16h bc = ld_frag_a(At + rc * 128 + hw * 8);       // mt=0, kk=0
#pragma unroll
  for (int kk = 0; kk < 4; ++kk) {
    const int kbB = kk * 32 + hw * 8;
    v16h a0n = a0, a1n = a1;
    if (kk < 3) {
      a0n = ld_frag_a(W1row0 + (kk + 1) * 32 + hw * 8);
      a1n = ld_frag_a(W1row1 + (kk + 1) * 32 + hw * 8);
    }
#pragma unroll
    for (int mt = 0; mt < 8; ++mt) {
      v16h bn = bc;
      if (mt < 7)       bn = ld_frag_a(At + ((mt + 1) * 16 + rc) * 128 + kbB);
      else if (kk < 3)  bn = ld_frag_a(At + rc * 128 + (kk + 1) * 32 + hw * 8);
      acc1[mt * 2 + 0] = WMMA_(a0, bc, acc1[mt * 2 + 0]);
      acc1[mt * 2 + 1] = WMMA_(a1, bc, acc1[mt * 2 + 1]);
      bc = bn;
    }
    a0 = a0n; a1 = a1n;
  }

  // bias + relu; acc element i = hidden index (wave*2+nt)*16 + hw*8 + i
  float bias1[2][8];
#pragma unroll
  for (int nt = 0; nt < 2; ++nt) {
    const float* bp = b1g + (wave * 2 + nt) * 16 + hw * 8;
    f4 lo = *(const f4*)bp, hi = *(const f4*)(bp + 4);
    bias1[nt][0] = lo.x; bias1[nt][1] = lo.y; bias1[nt][2] = lo.z; bias1[nt][3] = lo.w;
    bias1[nt][4] = hi.x; bias1[nt][5] = hi.y; bias1[nt][6] = hi.z; bias1[nt][7] = hi.w;
  }
  __syncthreads();                       // all waves done reading W1t / At
  // packed writeback: one b128 store per tile into H[m][k] (row-major)
#pragma unroll
  for (int mt = 0; mt < 8; ++mt)
#pragma unroll
    for (int nt = 0; nt < 2; ++nt) {
      v8f a = acc1[mt * 2 + nt];
      h8 hv;
#pragma unroll
      for (int i = 0; i < 8; ++i) {
        float t = a[i] + bias1[nt][i];
        hv[i] = (f16)(t > 0.f ? t : 0.f);
      }
      *(h8*)(Ht + (mt * 16 + rc) * 256 + (wave * 2 + nt) * 16 + hw * 8) = hv;
    }
  __syncthreads();

  // ===== Stage 2: O[m][j] = sum_k H[m][k] * W2t[j][k];  wave owns j-tile w
  v8f acc2[8];
#pragma unroll
  for (int i = 0; i < 8; ++i) acc2[i] = (v8f){};

  const f16* W2row = W2t + (wave * 16 + rc) * 256;
  v16h b2c = ld_frag_a(W2row + hw * 8);
  v16h a2c = ld_frag_a(Ht + rc * 256 + hw * 8);
#pragma unroll
  for (int kk = 0; kk < 8; ++kk) {
    const int kbA = kk * 32 + hw * 8;
    v16h b2n = b2c;
    if (kk < 7) b2n = ld_frag_a(W2row + (kk + 1) * 32 + hw * 8);
#pragma unroll
    for (int mt = 0; mt < 8; ++mt) {
      v16h a2n = a2c;
      if (mt < 7)       a2n = ld_frag_a(Ht + ((mt + 1) * 16 + rc) * 256 + kbA);
      else if (kk < 7)  a2n = ld_frag_a(Ht + rc * 256 + (kk + 1) * 32 + hw * 8);
      acc2[mt] = WMMA_(a2c, b2c, acc2[mt]);
      a2c = a2n;
    }
    b2c = b2n;
  }
  const float bias2 = b2g[wave * 16 + rc];
#pragma unroll
  for (int mt = 0; mt < 8; ++mt)
#pragma unroll
    for (int i = 0; i < 8; ++i) {
      long row = blockRow + mt * 16 + i + hw * 8;
      out[row * 128 + wave * 16 + rc] = acc2[mt][i] + bias2;
    }
}

// ---------------------------------------------------------------------------
// New edges: idx over (p, d, k) p-major; src = off(3-d) + (p>>d)*E + k%E,
// tgt = N0 + p*2E + k, attr = d+1.
// ---------------------------------------------------------------------------
__global__ void new_edges_kernel(int* __restrict__ out_edge,
                                 int* __restrict__ out_attr,
                                 int eold, int etot) {
  const long BE = 2L * E_;
  long idx = (long)blockIdx.x * blockDim.x + threadIdx.x;
  if (idx >= (long)P_ * D_ * BE) return;
  int  p   = (int)(idx / (D_ * BE));
  long rem = idx - (long)p * (D_ * BE);
  int  d   = (int)(rem / BE);            // 0..3 -> degree d+1
  int  k   = (int)(rem - (long)d * BE);
  int  e   = k & (E_ - 1);
  int  anc = E_ * ((1 << (3 - d)) - 1);  // _off(3-d)
  int  src = anc + (p >> d) * E_ + e;
  long tgt = (long)N0_ + (long)p * BE + k;
  out_edge[eold + idx]              = src;
  out_edge[(long)etot + eold + idx] = (int)tgt;
  out_attr[eold + idx]              = d + 1;
}

__global__ void new_event_kernel(int* __restrict__ out_event) {
  long j = (long)blockIdx.x * blockDim.x + threadIdx.x;
  if (j < (long)P_ * B_ * E_) out_event[N0_ + j] = (int)(j & (E_ - 1));
}

// ---------------------------------------------------------------------------
extern "C" void kernel_launch(void* const* d_in, const int* in_sizes, int n_in,
                              void* d_out, int out_size, void* d_ws, size_t ws_size,
                              hipStream_t stream) {
  const float* x  = (const float*)d_in[0];
  const int*   ei = (const int*)d_in[1];
  const int*   ea = (const int*)d_in[2];
  const int*   ev = (const int*)d_in[3];
  const float* gf = (const float*)d_in[4];
  const float* W1 = (const float*)d_in[5];
  const float* b1 = (const float*)d_in[6];
  const float* W2 = (const float*)d_in[7];
  const float* b2 = (const float*)d_in[8];

  const long EOLD = in_sizes[1] / 2;                 // 1,114,112
  const long ENEW = (long)P_ * D_ * 2 * E_;          // 2,097,152
  const long ETOT = EOLD + ENEW;                     // 3,211,264
  const long NEWROWS = (long)P_ * B_ * E_;           // 524,288

  float* out_x     = (float*)d_out;
  int*   out_edge  = (int*)d_out + (long)(N0_ + NEWROWS) * F_;
  int*   out_attr  = out_edge + 2 * ETOT;
  int*   out_event = out_attr + ETOT;

  // old segments: pure d2d copies
  hipMemcpyAsync(out_x, x, (size_t)N0_ * F_ * sizeof(float),
                 hipMemcpyDeviceToDevice, stream);
  hipMemcpyAsync(out_edge, ei, (size_t)EOLD * sizeof(int),
                 hipMemcpyDeviceToDevice, stream);                 // src row
  hipMemcpyAsync(out_edge + ETOT, ei + EOLD, (size_t)EOLD * sizeof(int),
                 hipMemcpyDeviceToDevice, stream);                 // tgt row
  hipMemcpyAsync(out_attr, ea, (size_t)EOLD * sizeof(int),
                 hipMemcpyDeviceToDevice, stream);
  hipMemcpyAsync(out_event, ev, (size_t)N0_ * sizeof(int),
                 hipMemcpyDeviceToDevice, stream);

  mlp_fused_kernel<<<ROWS_ / TM_, 256, 0, stream>>>(
      x, gf, W1, b1, W2, b2, out_x + (long)N0_ * F_);

  new_edges_kernel<<<(int)((ENEW + 255) / 256), 256, 0, stream>>>(
      out_edge, out_attr, (int)EOLD, (int)ETOT);
  new_event_kernel<<<(int)((NEWROWS + 255) / 256), 256, 0, stream>>>(out_event);
}
